// TinyInr_86964497809978
// MI455X (gfx1250) — compile-verified, matrix operand order checked by reading, not run on hardware
//
#include <hip/hip_runtime.h>
#include <hip/hip_bf16.h>

typedef _Float16 h8   __attribute__((ext_vector_type(8)));
typedef _Float16 v16h __attribute__((ext_vector_type(16)));
typedef float    v8f  __attribute__((ext_vector_type(8)));
typedef int      v4i  __attribute__((ext_vector_type(4)));

#define N_LEVELS    16
#define LOG2_T      19
#define T_SIZE      (1u << LOG2_T)
#define WIDTH       128
#define NPTS        2097152
#define PTS_PER_BLK 128
#define STRIDE      136          // halfs per act row: 272B, 16B aligned, bank-skewed
#define WBUF_HALFS  16384        // one 128x128 f16 weight layer in fragment order

// workspace layout in halfs: [w_in frags][w_hid frags][w_out frags]
#define WIN_OFF   0
#define WIN_CNT   4096           // 8 frags (ks=0, ct=0..7) * 512
#define WHID_OFF  4096
#define WHID_CNT  (7 * 16384)    // 7 layers * 32 frags * 512
#define WOUT_OFF  (WHID_OFF + WHID_CNT)   // 118784
#define WOUT_CNT  2048           // 4 frags (ks=0..3) * 512, N padded 3->16
#define WTOT      (WOUT_OFF + WOUT_CNT)   // 120832

// ---------------- async-to-LDS plumbing (guarded; falls back to sync copy) ---
#if defined(__gfx1250__) && __has_builtin(__builtin_amdgcn_global_load_async_to_lds_b128)
#define USE_ASYNC 1
#else
#define USE_ASYNC 0
#endif

#if defined(__gfx1250__) && __has_builtin(__builtin_amdgcn_s_wait_asynccnt)
#define WAIT_ASYNC(n) __builtin_amdgcn_s_wait_asynccnt(n)
#elif defined(__gfx1250__)
#define WAIT_ASYNC(n) asm volatile("s_wait_asynccnt %0" ::"n"(n) : "memory")
#else
#define WAIT_ASYNC(n)
#endif

#if USE_ASYNC
typedef __attribute__((address_space(1))) v4i* gvec_t;   // global int4*
typedef __attribute__((address_space(3))) v4i* lvec_t;   // LDS int4*
#endif

__device__ __forceinline__ void stage_weights(const _Float16* __restrict__ gsrc,
                                              _Float16* __restrict__ ldst,
                                              int tid, int nhalfs) {
#if USE_ASYNC
  int nvec = nhalfs >> 3;                      // 16B chunks
  for (int i = tid; i < nvec; i += 256) {
    __builtin_amdgcn_global_load_async_to_lds_b128(
        (gvec_t)(gsrc + i * 8), (lvec_t)(ldst + i * 8), 0, 0);
  }
#else
  const uint4* s = (const uint4*)gsrc;
  uint4* d = (uint4*)ldst;
  for (int i = tid; i < (nhalfs >> 3); i += 256) d[i] = s[i];
#endif
}

// ---------------------------------------------------------------------------
// Convert f32 weights to f16, pre-swizzled into WMMA B-fragment order:
// frag(ks,ct): lane holds column ct*16+(lane&15); element e -> K offset
// (e&7) + (e>=8 ? 16 : 0) + ((lane&16) ? 8 : 0)   (ISA 16-bit B layout)
// ---------------------------------------------------------------------------
__global__ void convert_weights(const float* __restrict__ w_in,
                                const float* __restrict__ w_hid,
                                const float* __restrict__ w_out,
                                _Float16* __restrict__ ws16) {
  int tid = blockIdx.x * blockDim.x + threadIdx.x;
  if (tid >= WTOT) return;
  int lane = (tid >> 4) & 31;
  int e    = tid & 15;
  int kk   = (e & 7) + ((e >> 3) << 4) + ((lane & 16) ? 8 : 0);  // 0..31 in K-block
  float v;
  if (tid < WHID_OFF) {                       // w_in [32 x 128], K=32 (ks==0)
    int ct  = tid >> 9;
    int col = ct * 16 + (lane & 15);
    v = w_in[kk * WIDTH + col];
  } else if (tid < WOUT_OFF) {                // w_hid [7][128][128]
    int local = tid - WHID_OFF;
    int layer = local >> 14;
    int r     = local & 16383;
    int frag  = r >> 9;
    int ks = frag >> 3, ct = frag & 7;
    int col = ct * 16 + (lane & 15);
    int k   = ks * 32 + kk;
    v = w_hid[((size_t)layer * WIDTH + k) * WIDTH + col];
  } else {                                    // w_out [128 x 3] padded to N=16
    int local = tid - WOUT_OFF;
    int ks = local >> 9;
    int n  = lane & 15;
    int k  = ks * 32 + kk;
    v = (n < 3) ? w_out[k * 3 + n] : 0.0f;
  }
  ws16[tid] = (_Float16)v;
}

// ---------------------------------------------------------------------------
__device__ __forceinline__ v16h cat8(h8 lo, h8 hi) {
  return __builtin_shufflevector(lo, hi, 0,1,2,3,4,5,6,7,8,9,10,11,12,13,14,15);
}

__global__ void __launch_bounds__(256)
tiny_inr_fused(const float* __restrict__ coords,
               const float* __restrict__ table,
               const _Float16* __restrict__ wfrag,
               float* __restrict__ out) {
  extern __shared__ _Float16 smem[];
  _Float16* bufA = smem;                               // 128 x STRIDE
  _Float16* bufB = smem + PTS_PER_BLK * STRIDE;        // 128 x STRIDE
  _Float16* wb0  = smem + 2 * PTS_PER_BLK * STRIDE;    // weight buffer 0
  _Float16* wb1  = wb0 + WBUF_HALFS;                   // weight buffer 1
  _Float16* wb[2] = {wb0, wb1};

  const int tid  = threadIdx.x;
  const int lane = tid & 31;
  const int wv   = tid >> 5;            // 8 waves -> 8 row tiles of 16
  const int blk  = blockIdx.x;

  // kick off async staging of w_in (wb0) and hid layer 0 (wb1); overlaps encode
  stage_weights(wfrag + WIN_OFF, wb0, tid, WIN_CNT);
  stage_weights(wfrag + WHID_OFF, wb1, tid, WBUF_HALFS);

  // ---- hash-grid encoding: 2 threads per point, 8 levels each -> bufA (K=32)
  {
    int p  = tid >> 1;
    int l0 = (tid & 1) * 8;
    float cx = coords[((size_t)blk * PTS_PER_BLK + p) * 2 + 0];
    float cy = coords[((size_t)blk * PTS_PER_BLK + p) * 2 + 1];
    #pragma unroll
    for (int li = 0; li < 8; ++li) {
      int l = l0 + li;
      unsigned res = 16u << l;
      float scale = (float)(res - 1u);
      float px = cx * scale + 0.5f, py = cy * scale + 0.5f;
      float fx = floorf(px), fy = floorf(py);
      float wx = px - fx, wy = py - fy;
      unsigned ix = (unsigned)(int)fx, iy = (unsigned)(int)fy;
      const float* tl = table + (size_t)l * (T_SIZE * 2);
      bool dense = (l < 6);                       // res^2 <= T for l<=5
      float f0 = 0.f, f1 = 0.f;
      #pragma unroll
      for (int c = 0; c < 4; ++c) {
        unsigned dx = (unsigned)(c >> 1), dy = (unsigned)(c & 1);
        unsigned gx = ix + dx, gy = iy + dy;
        unsigned idx;
        if (dense) {
          unsigned m = res - 1u;
          gx = gx < m ? gx : m;
          gy = gy < m ? gy : m;
          idx = gx + gy * res;
        } else {
          idx = (gx ^ (gy * 2654435761u)) & (T_SIZE - 1u);
        }
        float w = (dx ? wx : 1.f - wx) * (dy ? wy : 1.f - wy);
        f0 += w * tl[(size_t)idx * 2 + 0];
        f1 += w * tl[(size_t)idx * 2 + 1];
      }
      bufA[p * STRIDE + l * 2 + 0] = (_Float16)f0;
      bufA[p * STRIDE + l * 2 + 1] = (_Float16)f1;
    }
  }
  WAIT_ASYNC(0);          // w_in + hid0 staged (own wave's copies)
  __syncthreads();        // ...and everyone else's; encode also complete

  // one layer: dst[128x128] = relu(src[128xK] @ W), W fragments in wbuf
  auto layer = [&](const _Float16* aSrc, _Float16* aDst, const _Float16* wbuf,
                   int nks) {
    const int rt   = wv;
    const int rowl = lane & 15;
    const int row  = rt * 16 + rowl;
    const int kofs = (lane & 16) ? 8 : 0;
    v8f acc[8];
    #pragma unroll
    for (int ct = 0; ct < 8; ++ct) acc[ct] = v8f{0.f,0.f,0.f,0.f,0.f,0.f,0.f,0.f};
    for (int ks = 0; ks < nks; ++ks) {
      int kb = ks * 32 + kofs;
      h8 alo = *(const h8*)(aSrc + row * STRIDE + kb);
      h8 ahi = *(const h8*)(aSrc + row * STRIDE + kb + 16);
      v16h a = cat8(alo, ahi);
      v16h bf[8];
      #pragma unroll
      for (int ct = 0; ct < 8; ++ct) {
        const h8* bp = (const h8*)(wbuf + (ks * 8 + ct) * 512 + lane * 16);
        bf[ct] = cat8(bp[0], bp[1]);
      }
      #pragma unroll
      for (int ct = 0; ct < 8; ++ct) {
        acc[ct] = __builtin_amdgcn_wmma_f32_16x16x32_f16(
            false, a, false, bf[ct], (short)0, acc[ct], false, false);
      }
    }
    const int mofs = (lane & 16) ? 8 : 0;
    #pragma unroll
    for (int ct = 0; ct < 8; ++ct) {
      #pragma unroll
      for (int v = 0; v < 8; ++v) {
        float x = acc[ct][v];
        aDst[(rt * 16 + v + mofs) * STRIDE + ct * 16 + rowl] =
            (_Float16)(x > 0.f ? x : 0.f);
      }
    }
  };

  // layer 0: enc(K=32) @ w_in (wb0) -> bufB
  layer(bufA, bufB, wb0, 1);

  // 7 hidden layers; weights for hid[hl] live in wb[(hl+1)&1]
  const _Float16* src = bufB;
  _Float16* dst = bufA;
  for (int hl = 0; hl < 7; ++hl) {
    __syncthreads();                          // buffer wb[hl&1] is now free
    if (hl < 6) {
      stage_weights(wfrag + WHID_OFF + (hl + 1) * WBUF_HALFS, wb[hl & 1],
                    tid, WBUF_HALFS);         // prefetch next layer (8 instrs)
      WAIT_ASYNC(8);                          // current layer's copies done
    } else {
      stage_weights(wfrag + WOUT_OFF, wb[0], tid, WOUT_CNT);  // 1 instr
      WAIT_ASYNC(1);
    }
    __syncthreads();                          // cross-wave visibility
    layer(src, dst, wb[(hl + 1) & 1], 4);
    const _Float16* t = src; src = dst; dst = (_Float16*)t;
  }
  // final activations in bufA (== src); w_out frags landing in wb0

  WAIT_ASYNC(0);
  __syncthreads();

  // ---- output layer: [128x128] @ w_out(padded 128x16), write 3 cols
  {
    const int rt   = wv;
    const int rowl = lane & 15;
    const int row  = rt * 16 + rowl;
    const int kofs = (lane & 16) ? 8 : 0;
    v8f acc = v8f{0.f,0.f,0.f,0.f,0.f,0.f,0.f,0.f};
    #pragma unroll
    for (int ks = 0; ks < 4; ++ks) {
      int kb = ks * 32 + kofs;
      h8 alo = *(const h8*)(src + row * STRIDE + kb);
      h8 ahi = *(const h8*)(src + row * STRIDE + kb + 16);
      v16h a = cat8(alo, ahi);
      const h8* bp = (const h8*)(wb0 + ks * 512 + lane * 16);
      v16h b = cat8(bp[0], bp[1]);
      acc = __builtin_amdgcn_wmma_f32_16x16x32_f16(
          false, a, false, b, (short)0, acc, false, false);
    }
    const int n = lane & 15;
    const int mofs = (lane & 16) ? 8 : 0;
    if (n < 3) {
      #pragma unroll
      for (int v = 0; v < 8; ++v) {
        out[((size_t)blk * PTS_PER_BLK + rt * 16 + v + mofs) * 3 + n] = acc[v];
      }
    }
  }
}

// ---------------------------------------------------------------------------
extern "C" void kernel_launch(void* const* d_in, const int* in_sizes, int n_in,
                              void* d_out, int out_size, void* d_ws, size_t ws_size,
                              hipStream_t stream) {
  const float* coords = (const float*)d_in[0];
  const float* table  = (const float*)d_in[1];
  const float* w_in   = (const float*)d_in[2];
  const float* w_hid  = (const float*)d_in[3];
  const float* w_out  = (const float*)d_in[4];
  _Float16* ws16 = (_Float16*)d_ws;
  float* out = (float*)d_out;

  convert_weights<<<(WTOT + 255) / 256, 256, 0, stream>>>(w_in, w_hid, w_out, ws16);

  size_t smem_bytes = (size_t)(2 * PTS_PER_BLK * STRIDE + 2 * WBUF_HALFS) * sizeof(_Float16);
  tiny_inr_fused<<<NPTS / PTS_PER_BLK, 256, smem_bytes, stream>>>(coords, table, ws16, out);
}
